// Decoder_1108101563170
// MI455X (gfx1250) — compile-verified, matrix operand order checked by reading, not run on hardware
//
#include <hip/hip_runtime.h>
#include <hip/hip_bf16.h>

// CDNA5 wave32 WMMA fragment types
typedef __attribute__((ext_vector_type(16))) __bf16 v16bf;
typedef __attribute__((ext_vector_type(8)))  __bf16 v8bf;
typedef __attribute__((ext_vector_type(8)))  float  v8f;

constexpr int BB = 64;    // batch
constexpr int TT = 128;   // sequence length
constexpr int SS = 64;    // context length
constexpr int HH = 512;   // hidden
constexpr int EE = 512;   // embedding
constexpr int NWG = 32;   // persistent workgroups (one 16-wide hidden slice each)
constexpr int NK0 = 48;   // layer0 K/32  (K = E + H + H = 1536)
constexpr int NK1 = 32;   // layer1 K/32  (K = H + H = 1024)

// LDS: weights pre-packed in WMMA B-fragment order.
// fragment(ks, gate) = 1024B: [granule q=0|1][lane 0..31] x 16B  -> conflict-free ds_load_b128
constexpr int LDS_L0    = 0;
constexpr int LDS_L1    = NK0 * 4 * 1024;            // 196608
constexpr int LDS_TOTAL = LDS_L1 + NK1 * 4 * 1024;   // 327680 = 320 KB (full WGP LDS)

__device__ __forceinline__ float sigf(float x) { return 1.0f / (1.0f + __expf(-x)); }

union fragU { v16bf v; v8bf h[2]; };
union packU { __bf16 e[8]; v8bf v; };

// ---------------- prep kernels ----------------

__global__ __launch_bounds__(32) void k_zero(int* p) {
  if (threadIdx.x == 0 && blockIdx.x == 0) *p = 0;
}

__global__ __launch_bounds__(256) void k_f32_to_bf16(const float* __restrict__ src,
                                                     __bf16* __restrict__ dst, int n) {
  for (int i = blockIdx.x * blockDim.x + threadIdx.x; i < n; i += gridDim.x * blockDim.x)
    dst[i] = (__bf16)src[i];
}

// emb[t][b][e] = bf16(W_emb[tokens[b][t]][e])  (padding row 0 already zeroed upstream)
__global__ __launch_bounds__(256) void k_embed(const int* __restrict__ tokens,
                                               const float* __restrict__ W_emb,
                                               __bf16* __restrict__ emb) {
  const int n = TT * BB * EE;
  for (int i = blockIdx.x * blockDim.x + threadIdx.x; i < n; i += gridDim.x * blockDim.x) {
    int e  = i & (EE - 1);
    int tb = i >> 9;
    int b  = tb & (BB - 1);
    int t  = tb >> 6;
    int tok = tokens[b * TT + t];
    emb[i] = (__bf16)W_emb[(size_t)tok * EE + e];
  }
}

// ---------------- persistent decoder kernel ----------------
// grid = 32 workgroups (hidden slice of 16 each), block = 4 waves (M-tiles of 16).
// Requires all 32 workgroups co-resident (1 WGP each at 320KB LDS) -- trivially true on MI455X.

__global__ __launch_bounds__(128) void k_persist(
    const float* __restrict__ W_ih0, const float* __restrict__ W_hh0,
    const float* __restrict__ b_ih0, const float* __restrict__ b_hh0,
    const float* __restrict__ W_ih1, const float* __restrict__ W_hh1,
    const float* __restrict__ b_ih1, const float* __restrict__ b_hh1,
    const float* __restrict__ h0in, const float* __restrict__ c0in,
    const float* __restrict__ initout, const float* __restrict__ ctx,
    const __bf16* __restrict__ EMB,    // [T][B][E] bf16
    const __bf16* __restrict__ WINB,   // [512][512] bf16
    const __bf16* __restrict__ WOUTB,  // [512][1024] bf16
    __bf16* __restrict__ hst0,         // [2][64*512] ping-pong layer0 h
    __bf16* __restrict__ hst1,         // [2][64*512] ping-pong layer1 h
    __bf16* __restrict__ ofeed,        // [2][64*512] ping-pong out-feed
    __bf16* __restrict__ h1buf, __bf16* __restrict__ h2buf,
    __bf16* __restrict__ wctxb,        // [64*512] bf16 weighted context
    float* __restrict__ QB,            // [64*512] f32 attention query
    float* __restrict__ outp,          // [B][T][H] f32 output
    int* __restrict__ barcnt) {
  extern __shared__ char smem[];
  const int tid   = threadIdx.x;
  const int s     = blockIdx.x;        // hidden slice
  const int wave  = tid >> 5;
  const int lane  = tid & 31;
  const int lhalf = lane >> 4;
  const int l15   = lane & 15;
  const int nloc  = s * 16 + l15;      // output column owned by this lane
  const int arow  = wave * 16 + l15;   // A-fragment row for this lane
  int done = 0;

  // ---- device-wide barrier: monotonic counter, acq/rel at AGENT scope ----
  auto gbar = [&]() {
    __syncthreads();
    done += NWG;
    if (tid == 0) {
      __hip_atomic_fetch_add(barcnt, 1, __ATOMIC_ACQ_REL, __HIP_MEMORY_SCOPE_AGENT);
      while (__hip_atomic_load(barcnt, __ATOMIC_ACQUIRE, __HIP_MEMORY_SCOPE_AGENT) < done)
        __builtin_amdgcn_s_sleep(1);
    }
    __syncthreads();
  };

  // ---- stage this slice's gate weights into LDS in B-fragment order (once) ----
  // granule (ks, gate, q, ln): B-frag column j = ln&15 (global row gate*512+s*16+j),
  // K = ks*32 + (ln>>4)*16 + q*8 .. +7
  for (int idx = tid; idx < NK0 * 4 * 64; idx += 128) {
    const int g64 = idx & 63, gate = (idx >> 6) & 3, ks = idx >> 8;
    const int q = g64 >> 5, ln = g64 & 31;
    const int row = gate * HH + s * 16 + (ln & 15);
    const int kst = ks * 32 + (ln >> 4) * 16 + q * 8;
    packU pk;
#pragma unroll
    for (int j = 0; j < 8; j++) {
      const int col = kst + j;
      const float v = (col < 1024) ? W_ih0[(size_t)row * 1024 + col]
                                   : W_hh0[(size_t)row * 512 + (col - 1024)];
      pk.e[j] = (__bf16)v;
    }
    *(v8bf*)(smem + LDS_L0 + (ks * 4 + gate) * 1024 + q * 512 + ln * 16) = pk.v;
  }
  for (int idx = tid; idx < NK1 * 4 * 64; idx += 128) {
    const int g64 = idx & 63, gate = (idx >> 6) & 3, ks = idx >> 8;
    const int q = g64 >> 5, ln = g64 & 31;
    const int row = gate * HH + s * 16 + (ln & 15);
    const int kst = ks * 32 + (ln >> 4) * 16 + q * 8;
    packU pk;
#pragma unroll
    for (int j = 0; j < 8; j++) {
      const int col = kst + j;
      const float v = (col < 512) ? W_ih1[(size_t)row * 512 + col]
                                  : W_hh1[(size_t)row * 512 + (col - 512)];
      pk.e[j] = (__bf16)v;
    }
    *(v8bf*)(smem + LDS_L1 + (ks * 4 + gate) * 1024 + q * 512 + ln * 16) = pk.v;
  }

  // ---- init recurrent state; cell state lives in VGPRs for the whole sequence ----
  float c0r[8], c1r[8];
#pragma unroll
  for (int r = 0; r < 8; r++) {
    const int m = wave * 16 + 8 * lhalf + r;   // WMMA C-layout row
    const int idx = m * HH + nloc;
    c0r[r] = c0in[idx];
    c1r[r] = c0in[BB * HH + idx];
    hst0[idx]  = (__bf16)h0in[idx];
    hst1[idx]  = (__bf16)h0in[BB * HH + idx];
    ofeed[idx] = (__bf16)initout[idx];
  }
  float bs0[4], bs1[4];
#pragma unroll
  for (int g = 0; g < 4; g++) {
    bs0[g] = b_ih0[g * HH + nloc] + b_hh0[g * HH + nloc];
    bs1[g] = b_ih1[g * HH + nloc] + b_hh1[g * HH + nloc];
  }
  gbar();

  // ---- fused LSTM layer: WMMA gates from LDS weights + in-register cell update ----
  auto lstm_phase = [&](const char* Wl, int NK, const __bf16* a0, const __bf16* a1,
                        const __bf16* a2, const float* bs, float* cr,
                        __bf16* hA, __bf16* hB) {
    v8f ai = {}, af = {}, ag = {}, ao = {};
    for (int ks = 0; ks < NK; ks++) {
      const __bf16* seg = (ks < 16) ? a0 : ((ks < 32) ? a1 : a2);  // 512-col A segments
      const int ko = ((ks * 32) & 511) + lhalf * 16;
      v16bf a = *(const v16bf*)(seg + (size_t)arow * 512 + ko);
      const char* fb = Wl + ks * 4096;
      fragU w0, w1, w2, w3;
      w0.h[0] = *(const v8bf*)(fb + 0    + lane * 16);
      w0.h[1] = *(const v8bf*)(fb + 512  + lane * 16);
      w1.h[0] = *(const v8bf*)(fb + 1024 + lane * 16);
      w1.h[1] = *(const v8bf*)(fb + 1536 + lane * 16);
      w2.h[0] = *(const v8bf*)(fb + 2048 + lane * 16);
      w2.h[1] = *(const v8bf*)(fb + 2560 + lane * 16);
      w3.h[0] = *(const v8bf*)(fb + 3072 + lane * 16);
      w3.h[1] = *(const v8bf*)(fb + 3584 + lane * 16);
      ai = __builtin_amdgcn_wmma_f32_16x16x32_bf16(false, a, false, w0.v, (short)0, ai, false, false);
      af = __builtin_amdgcn_wmma_f32_16x16x32_bf16(false, a, false, w1.v, (short)0, af, false, false);
      ag = __builtin_amdgcn_wmma_f32_16x16x32_bf16(false, a, false, w2.v, (short)0, ag, false, false);
      ao = __builtin_amdgcn_wmma_f32_16x16x32_bf16(false, a, false, w3.v, (short)0, ao, false, false);
    }
#pragma unroll
    for (int r = 0; r < 8; r++) {
      const int m = wave * 16 + 8 * lhalf + r;
      const int idx = m * HH + nloc;
      const float iv = sigf(ai[r] + bs[0]);
      const float fv = sigf(af[r] + bs[1]);
      const float gv = tanhf(ag[r] + bs[2]);
      const float ov = sigf(ao[r] + bs[3]);
      const float c2 = fv * cr[r] + iv * gv;
      cr[r] = c2;
      const float h = ov * tanhf(c2);
      hA[idx] = (__bf16)h;
      hB[idx] = (__bf16)h;
    }
  };

  // ---- projection GEMM: D[64,512] = act(A @ W^T), W bf16 from L2 ----
  auto proj_phase = [&](const __bf16* a0, const __bf16* a1, const __bf16* W, int NK,
                        float* oF, int oStride, __bf16* oB, bool dotanh) {
    const __bf16* Wr = W + (size_t)nloc * (NK * 32);
    v8f acc = {};
    for (int ks = 0; ks < NK; ks++) {
      const __bf16* seg = (ks < 16) ? a0 : a1;
      const int ko = ((ks * 32) & 511) + lhalf * 16;
      v16bf a = *(const v16bf*)(seg + (size_t)arow * 512 + ko);
      v16bf w = *(const v16bf*)(Wr + ks * 32 + lhalf * 16);
      acc = __builtin_amdgcn_wmma_f32_16x16x32_bf16(false, a, false, w, (short)0, acc, false, false);
    }
#pragma unroll
    for (int r = 0; r < 8; r++) {
      const int m = wave * 16 + 8 * lhalf + r;
      float v = acc[r];
      if (dotanh) v = tanhf(v);
      if (oF) oF[(size_t)m * oStride + nloc] = v;
      if (oB) oB[m * HH + nloc] = (__bf16)v;
    }
  };

  // ---- sequential decode: entire time loop inside one kernel ----
  for (int t = 0; t < TT; t++) {
    const int p = t & 1;
    // layer 0: x = [emb_t | out_feed | h0_prev]; h1 -> h1buf and h0[1-p]
    lstm_phase(smem + LDS_L0, NK0, EMB + (size_t)t * BB * EE, ofeed + p * BB * HH,
               hst0 + p * BB * HH, bs0, c0r, h1buf, hst0 + (1 - p) * BB * HH);
    gbar();
    // layer 1: x = [h1 | h1_prev]; h2 -> h2buf and h1[1-p]
    lstm_phase(smem + LDS_L1, NK1, h1buf, hst1 + p * BB * HH, nullptr,
               bs1, c1r, h2buf, hst1 + (1 - p) * BB * HH);
    gbar();
    // q = h2 @ W_in^T
    proj_phase(h2buf, nullptr, WINB, 16, QB, HH, nullptr, false);
    gbar();
    // attention: one batch per wave (waves 0,1), softmax via wave shuffles (no LDS)
    if (wave < 2) {
      const int b = s * 2 + wave;
      const float* qb = QB + (size_t)b * HH;
      const float* cb = ctx + (size_t)b * SS * HH;
      const float* r0 = cb + (size_t)lane * HH;
      const float* r1 = cb + (size_t)(lane + 32) * HH;
      float d0 = 0.f, d1 = 0.f;
      for (int h = 0; h < HH; h++) { const float qv = qb[h]; d0 += r0[h] * qv; d1 += r1[h] * qv; }
      float mx = fmaxf(d0, d1);
      for (int o = 16; o > 0; o >>= 1) mx = fmaxf(mx, __shfl_xor(mx, o, 32));
      float e0 = __expf(d0 - mx), e1 = __expf(d1 - mx);
      float sm = e0 + e1;
      for (int o = 16; o > 0; o >>= 1) sm += __shfl_xor(sm, o, 32);
      const float inv = 1.0f / sm;
      e0 *= inv; e1 *= inv;
      float acc[16];
#pragma unroll
      for (int k = 0; k < 16; k++) acc[k] = 0.f;
      for (int j = 0; j < 32; j++) {
        const float pj  = __shfl(e0, j, 32);
        const float pj2 = __shfl(e1, j, 32);
        const float* rj  = cb + (size_t)j * HH;
        const float* rj2 = cb + (size_t)(j + 32) * HH;
#pragma unroll
        for (int k = 0; k < 16; k++) {
          const int hc = lane + 32 * k;
          acc[k] += pj * rj[hc] + pj2 * rj2[hc];
        }
      }
#pragma unroll
      for (int k = 0; k < 16; k++) wctxb[(size_t)b * HH + lane + 32 * k] = (__bf16)acc[k];
    }
    gbar();
    // out = tanh([wctx | h2] @ W_out^T) -> d_out[:,t,:] and out-feed[1-p]
    proj_phase(wctxb, h2buf, WOUTB, 32, outp + (size_t)t * HH, TT * HH,
               ofeed + (1 - p) * BB * HH, true);
    gbar();
  }
}

// ---------------- host orchestration ----------------

extern "C" void kernel_launch(void* const* d_in, const int* in_sizes, int n_in,
                              void* d_out, int out_size, void* d_ws, size_t ws_size,
                              hipStream_t stream) {
  (void)in_sizes; (void)n_in; (void)out_size; (void)ws_size;
  const int*   tokens   = (const int*)d_in[0];
  const float* context  = (const float*)d_in[1];
  const float* h0       = (const float*)d_in[2];
  const float* c0       = (const float*)d_in[3];
  const float* init_out = (const float*)d_in[4];
  const float* W_emb    = (const float*)d_in[5];
  const float* W_ih0    = (const float*)d_in[6];
  const float* W_hh0    = (const float*)d_in[7];
  const float* b_ih0    = (const float*)d_in[8];
  const float* b_hh0    = (const float*)d_in[9];
  const float* W_ih1    = (const float*)d_in[10];
  const float* W_hh1    = (const float*)d_in[11];
  const float* b_ih1    = (const float*)d_in[12];
  const float* b_hh1    = (const float*)d_in[13];
  const float* W_in     = (const float*)d_in[14];
  const float* W_out    = (const float*)d_in[15];
  float* outp = (float*)d_out;

  char* ws = (char*)d_ws;
  size_t off = 0;
  auto alloc = [&](size_t bytes) -> void* {
    void* p = ws + off;
    off = (off + bytes + 255) & ~(size_t)255;
    return p;
  };
  int*    BARC  = (int*)alloc(256);
  __bf16* WINB  = (__bf16*)alloc((size_t)512 * 512 * 2);
  __bf16* WOUTB = (__bf16*)alloc((size_t)512 * 1024 * 2);
  __bf16* EMB   = (__bf16*)alloc((size_t)TT * BB * EE * 2);
  __bf16* HST0  = (__bf16*)alloc((size_t)2 * BB * HH * 2);
  __bf16* HST1  = (__bf16*)alloc((size_t)2 * BB * HH * 2);
  __bf16* OFEED = (__bf16*)alloc((size_t)2 * BB * HH * 2);
  __bf16* H1B   = (__bf16*)alloc((size_t)BB * HH * 2);
  __bf16* H2B   = (__bf16*)alloc((size_t)BB * HH * 2);
  __bf16* WCTX  = (__bf16*)alloc((size_t)BB * HH * 2);
  float*  QB    = (float*)alloc((size_t)BB * HH * 4);

  k_zero<<<1, 32, 0, stream>>>(BARC);
  k_f32_to_bf16<<<1024, 256, 0, stream>>>(W_in, WINB, 512 * 512);
  k_f32_to_bf16<<<2048, 256, 0, stream>>>(W_out, WOUTB, 512 * 1024);
  k_embed<<<8192, 256, 0, stream>>>(tokens, W_emb, EMB);

  k_persist<<<NWG, 128, LDS_TOTAL, stream>>>(
      W_ih0, W_hh0, b_ih0, b_hh0, W_ih1, W_hh1, b_ih1, b_hh1,
      h0, c0, init_out, context, EMB, WINB, WOUTB,
      HST0, HST1, OFEED, H1B, H2B, WCTX, QB, outp, BARC);
}
